// GCN_42872363548952
// MI455X (gfx1250) — matrix operand order repaired and checked by reading, not compiled, not run on hardware
//
#include <hip/hip_runtime.h>

// CDNA5 / gfx1250: wave32, WMMA 16x16x32 bf16 -> f32 accumulate
typedef __attribute__((ext_vector_type(16))) __bf16 bf16x16;
typedef __attribute__((ext_vector_type(8)))  __bf16 bf16x8;
typedef __attribute__((ext_vector_type(8)))  float  f32x8;

// ---------------- degree / dinv ----------------
__global__ void gcn_deg_init(float* __restrict__ deg, int n) {
  int i = blockIdx.x * blockDim.x + threadIdx.x;
  if (i < n) deg[i] = 1.0f;                 // self-loop contribution
}

__global__ void gcn_deg_accum(const int* __restrict__ dst, float* __restrict__ deg, int E) {
  int e = blockIdx.x * blockDim.x + threadIdx.x;
  if (e < E) atomicAdd(&deg[dst[e]], 1.0f);
}

__global__ void gcn_dinv(float* __restrict__ deg, int n) {
  int i = blockIdx.x * blockDim.x + threadIdx.x;
  if (i < n) deg[i] = rsqrtf(deg[i]);       // deg >= 1 always (self-loops)
}

// ---------------- activation f32 -> padded bf16 (optional fused ReLU) ----------------
__global__ void gcn_a_to_bf16(const float* __restrict__ A, __bf16* __restrict__ Abf,
                              int M, int K, int Kpad, int doRelu) {
  size_t idx = (size_t)blockIdx.x * blockDim.x + threadIdx.x;
  if (idx >= (size_t)M * Kpad) return;
  int m = (int)(idx / Kpad);
  int k = (int)(idx - (size_t)m * Kpad);
  float v = 0.0f;
  if (k < K) {
    v = A[(size_t)m * K + k];
    if (doRelu) v = fmaxf(v, 0.0f);
  }
  Abf[idx] = (__bf16)v;
}

// ---------------- weight transpose + f32->bf16 (zero-padded K) ----------------
// W: [K, Ncol] f32 row-major  ->  Wt: [Ncol, Kpad] bf16 row-major
__global__ void gcn_w_transpose_bf16(const float* __restrict__ W, __bf16* __restrict__ Wt,
                                     int K, int Ncol, int Kpad) {
  int idx = blockIdx.x * blockDim.x + threadIdx.x;
  if (idx >= Ncol * Kpad) return;
  int n = idx / Kpad;
  int k = idx - n * Kpad;
  float v = (k < K) ? W[(size_t)k * Ncol + n] : 0.0f;
  Wt[idx] = (__bf16)v;
}

// ---------------- WMMA GEMM: C[M,Ncol] = Abf[M,KP] @ Bt^T (+bias) ----------------
// Abf: bf16 [M,KP]; Bt: bf16 [Ncol,KP] (pre-transposed, zero-padded K).
// B staged in LDS once per block (shared by all 8 waves), zero-filled column tail.
// One wave32 per 16-row strip; TN column tiles in registers; A regs reused across TN WMMAs.
// Compile-time KP -> fully unrolled k-loop; EXEC all-ones; branch-free hot loop.
template<int TN, int KP>
__global__ __launch_bounds__(256)
void gcn_wmma_gemm(const __bf16* __restrict__ Abf, const __bf16* __restrict__ Bt,
                   const float* __restrict__ bias, float* __restrict__ C,
                   int M, int Ncol, int tilesM) {
  constexpr int ROWS = TN * 16;             // staged B rows (= padded Ncol)
  constexpr int CPR  = KP * 2 / 16;         // 16-byte chunks per B row
  __shared__ __align__(16) __bf16 Bs[ROWS * KP];

  // cooperative stage: global -> LDS, zero-fill rows >= Ncol
  for (int idx = threadIdx.x; idx < ROWS * CPR; idx += 256) {
    int r = idx / CPR, c = idx - r * CPR;
    uint4 v = {0u, 0u, 0u, 0u};
    if (r < Ncol) v = *(const uint4*)(Bt + (size_t)r * KP + c * 8);
    *(uint4*)(Bs + (size_t)r * KP + c * 8) = v;
  }
  __syncthreads();                          // all waves reach this before any early-out

  int gwave = (int)((blockIdx.x * blockDim.x + threadIdx.x) >> 5);
  if (gwave >= tilesM) return;              // wave-uniform
  int lane = threadIdx.x & 31;
  int half = lane >> 4;                     // 0: lanes 0-15, 1: lanes 16-31
  int mn   = lane & 15;

  int row  = gwave * 16 + mn;
  int rowc = row < M ? row : (M - 1);       // clamped: OOB rows never stored
  const __bf16* Arow = Abf + (size_t)rowc * KP;

  f32x8 acc[TN];
#pragma unroll
  for (int t = 0; t < TN; ++t) acc[t] = (f32x8){};

#pragma unroll
  for (int kb = 0; kb < KP; kb += 32) {
    // A lane layout (contiguous, matches B): j=0..15 -> K = kb + 16h + j
    bf16x16 a = *(const bf16x16*)(Arow + kb + half * 16);

#pragma unroll
    for (int t = 0; t < TN; ++t) {
      // B lane layout (col-major per lane): j=0..15 -> K = kb + 16h + j; from LDS (ds_load_b128 x2)
      bf16x16 b = *(const bf16x16*)(Bs + (size_t)(t * 16 + mn) * KP + kb + half * 16);
      acc[t] = __builtin_amdgcn_wmma_f32_16x16x32_bf16(
          /*neg_a=*/false, a, /*neg_b=*/false, b,
          /*c_mod=*/(short)0, acc[t], /*reuse_a=*/false, /*reuse_b=*/false);
    }
  }

  // C/D layout: VGPR i -> row gwave*16 + i + 8*half, col = t*16 + (lane&15)
#pragma unroll
  for (int t = 0; t < TN; ++t) {
    int coln = t * 16 + mn;
    if (coln < Ncol) {
      float bv = bias ? bias[coln] : 0.0f;
#pragma unroll
      for (int i = 0; i < 8; ++i) {
        int m = gwave * 16 + i + 8 * half;
        if (m < M) C[(size_t)m * Ncol + coln] = acc[t][i] + bv;
      }
    }
  }
}

// ---------------- aggregation ----------------
// out[n,f] = bias[f] + dinv[n]^2 * h[n,f]   (self-loop edge + bias)
__global__ void gcn_agg_init(const float* __restrict__ h, const float* __restrict__ dinv,
                             const float* __restrict__ bias, float* __restrict__ out,
                             int N, int F) {
  size_t idx = (size_t)blockIdx.x * blockDim.x + threadIdx.x;
  if (idx >= (size_t)N * F) return;
  int n = (int)(idx / F);
  int f = (int)(idx - (size_t)n * F);
  float di = dinv[n];
  out[idx] = bias[f] + di * di * h[idx];
}

// one wave32 per edge; lanes stride over features (coalesced gather + L2-resident f32 atomics)
__global__ void gcn_agg_edge(const float* __restrict__ h, const int* __restrict__ src,
                             const int* __restrict__ dst, const float* __restrict__ dinv,
                             float* __restrict__ out, int E, int F) {
  int e = (int)((blockIdx.x * blockDim.x + threadIdx.x) >> 5);
  if (e >= E) return;
  int lane = threadIdx.x & 31;
  int s = src[e], d = dst[e];
  float norm = dinv[s] * dinv[d];
  const float* hs = h + (size_t)s * F;
  float* od = out + (size_t)d * F;
  for (int f = lane; f < F; f += 32)
    atomicAdd(&od[f], hs[f] * norm);
}

// ---------------- host launch ----------------
extern "C" void kernel_launch(void* const* d_in, const int* in_sizes, int n_in,
                              void* d_out, int out_size, void* d_ws, size_t ws_size,
                              hipStream_t stream) {
  const int F_IN = 100, F_HID = 100, F_EMB = 64, N_CLS = 40;
  const float* x  = (const float*)d_in[0];
  const int*   ei = (const int*)  d_in[1];
  const float* W1 = (const float*)d_in[2];
  const float* b1 = (const float*)d_in[3];
  const float* W2 = (const float*)d_in[4];
  const float* b2 = (const float*)d_in[5];
  const float* Wc = (const float*)d_in[6];
  const float* bc = (const float*)d_in[7];
  float* out = (float*)d_out;

  const int N = in_sizes[0] / F_IN;
  const int E = in_sizes[1] / 2;
  const int* src = ei;        // edge_index[0]
  const int* dst = ei + E;    // edge_index[1]

  const int Kp128 = 128;      // K=100 -> 128
  const int Kp64  = 64;       // K=64  -> 64

  // workspace carve-out (floats, 256B-aligned chunks); buffers aliased across layers
  float* wsf = (float*)d_ws;
  auto au = [](size_t v) { return (v + 63) & ~(size_t)63; };
  size_t o = 0;
  float* deg  = wsf + o; o += au((size_t)N);               // deg -> dinv (in place)
  float* hbuf = wsf + o; o += au((size_t)N * F_HID);       // h1, later h2
  float* aggb = wsf + o; o += au((size_t)N * F_HID);       // agg1, later agg2
  __bf16* abf = (__bf16*)(wsf + o); o += au((size_t)N * Kp128 / 2); // xbf -> a1bf -> a2bf
  __bf16* Wt1 = (__bf16*)(wsf + o); o += au((size_t)F_HID * Kp128 / 2);
  __bf16* Wt2 = (__bf16*)(wsf + o); o += au((size_t)F_EMB * Kp128 / 2);
  __bf16* Wtc = (__bf16*)(wsf + o); o += au((size_t)N_CLS * Kp64 / 2);

  const int BT = 256;                        // 8 wave32 per block
  auto cdiv = [](long long a, long long b) { return (int)((a + b - 1) / b); };
  const int tilesM = cdiv(N, 16);
  const int gemmBlocks = cdiv(tilesM, 8);

  // ---- degrees ----
  gcn_deg_init <<<cdiv(N, BT), BT, 0, stream>>>(deg, N);
  gcn_deg_accum<<<cdiv(E, BT), BT, 0, stream>>>(dst, deg, E);
  gcn_dinv     <<<cdiv(N, BT), BT, 0, stream>>>(deg, N);

  // ---- layer 1: h1 = x @ W1 ; agg1 = D^-1/2 A D^-1/2 h1 + b1 ----
  gcn_a_to_bf16<<<cdiv((long long)N * Kp128, BT), BT, 0, stream>>>(x, abf, N, F_IN, Kp128, 0);
  gcn_w_transpose_bf16<<<cdiv((long long)F_HID * Kp128, BT), BT, 0, stream>>>(W1, Wt1, F_IN, F_HID, Kp128);
  gcn_wmma_gemm<7, 128><<<gemmBlocks, BT, 0, stream>>>(abf, Wt1, nullptr, hbuf, N, F_HID, tilesM);
  gcn_agg_init<<<cdiv((long long)N * F_HID, BT), BT, 0, stream>>>(hbuf, deg, b1, aggb, N, F_HID);
  gcn_agg_edge<<<cdiv(E, 8), BT, 0, stream>>>(hbuf, src, dst, deg, aggb, E, F_HID);

  // ---- layer 2: h2 = relu(agg1) @ W2 ; agg2 = D^-1/2 A D^-1/2 h2 + b2 ----
  gcn_a_to_bf16<<<cdiv((long long)N * Kp128, BT), BT, 0, stream>>>(aggb, abf, N, F_HID, Kp128, 1); // fused ReLU
  gcn_w_transpose_bf16<<<cdiv((long long)F_EMB * Kp128, BT), BT, 0, stream>>>(W2, Wt2, F_HID, F_EMB, Kp128);
  gcn_wmma_gemm<4, 128><<<gemmBlocks, BT, 0, stream>>>(abf, Wt2, nullptr, hbuf, N, F_EMB, tilesM);
  gcn_agg_init<<<cdiv((long long)N * F_EMB, BT), BT, 0, stream>>>(hbuf, deg, b2, aggb, N, F_EMB);
  gcn_agg_edge<<<cdiv(E, 8), BT, 0, stream>>>(hbuf, src, dst, deg, aggb, E, F_EMB);

  // ---- head: out = relu(agg2) @ Wc + bc ----
  gcn_a_to_bf16<<<cdiv((long long)N * Kp64, BT), BT, 0, stream>>>(aggb, abf, N, F_EMB, Kp64, 1);   // fused ReLU
  gcn_w_transpose_bf16<<<cdiv((long long)N_CLS * Kp64, BT), BT, 0, stream>>>(Wc, Wtc, F_EMB, N_CLS, Kp64);
  gcn_wmma_gemm<3, 64><<<gemmBlocks, BT, 0, stream>>>(abf, Wtc, bc, out, N, N_CLS, tilesM);
}
